// UnifiedRadialMLP_48782238548455
// MI455X (gfx1250) — compile-verified, matrix operand order checked by reading, not run on hardware
//
#include <hip/hip_runtime.h>

// Fused 3-stage MLP for MI455X (gfx1250), wave32 + WMMA bf16.
// N=8 branches, H=IN=OUT=128, E=65536. One workgroup = (branch n, 128-row e-tile).
// All intermediates stay in VGPRs/LDS; HBM traffic = read x + write out only.
// Output is staged through LDS so the HBM writes are fully coalesced b128 stores.

#define E_TOT 65536
#define HDIM  128

typedef __attribute__((ext_vector_type(16))) __bf16 v16bf;
typedef __attribute__((ext_vector_type(8)))  __bf16 v8bf;
typedef __attribute__((ext_vector_type(8)))  float  v8f;

// ---- load a 16x32 bf16 WMMA fragment from a row-major [*,128] bf16 LDS tile.
// A-matrix layout (ISA 7.12.2): lane L -> row (row0 + L%16); element e holds
// K = (e/8)*16 + (L/16)*8 + (e%8), i.e. two contiguous 8-element (16B) chunks.
// The B operand of out = act * W^T uses the identical gather on row-major W.
__device__ __forceinline__ v16bf load_frag(const __bf16* base, int row0, int kOff, int lane) {
  const int m    = lane & 15;
  const int half = lane >> 4;
  const __bf16* p = base + (row0 + m) * HDIM + kOff + half * 8;
  v16bf out;
  ((v8bf*)&out)[0] = *(const v8bf*)(p);
  ((v8bf*)&out)[1] = *(const v8bf*)(p + 16);
  return out;
}

// ---- 128x128x128 GEMM per workgroup; this wave computes rows [rowBase, rowBase+16).
__device__ __forceinline__ void gemm128(v8f acc[8], const __bf16* sA, const __bf16* sB,
                                        int rowBase, int lane) {
#pragma unroll
  for (int t = 0; t < 8; ++t)
#pragma unroll
    for (int r = 0; r < 8; ++r) acc[t][r] = 0.0f;

#pragma unroll
  for (int kb = 0; kb < 4; ++kb) {
    v16bf a = load_frag(sA, rowBase, kb * 32, lane);
#pragma unroll
    for (int t = 0; t < 8; ++t) {
      v16bf b = load_frag(sB, t * 16, kb * 32, lane);
      acc[t] = __builtin_amdgcn_wmma_f32_16x16x32_bf16(
          /*neg_a=*/false, a, /*neg_b=*/false, b,
          /*c_mod=*/(short)0, acc[t], /*reuse_a=*/false, /*reuse_b=*/false);
    }
  }
}

// ---- cooperative fp32 -> bf16 tile load (128x128) into LDS.
__device__ __forceinline__ void load_tile_bf16(__bf16* dst, const float* __restrict__ src, int tid) {
#pragma unroll
  for (int it = 0; it < 16; ++it) {
    const int i = (it * 256 + tid) * 4;
    const float4 v = *(const float4*)(src + i);
    dst[i + 0] = (__bf16)v.x;
    dst[i + 1] = (__bf16)v.y;
    dst[i + 2] = (__bf16)v.z;
    dst[i + 3] = (__bf16)v.w;
  }
}

// ---- bias + LayerNorm(H=128) + SiLU, fully in registers.
// C/D layout: VGPR r, lane L -> row (r + 8*(L/16)), col (L%16) of each 16x16 tile.
// A full H-row lives in 8 acc registers x the 16 lanes of one half-wave, so the
// LN reduction is 8 in-lane adds + 4 shfl_xor steps (masks 1,2,4,8 stay in-half).
__device__ __forceinline__ void ln_silu(v8f acc[8], const float* __restrict__ bias,
                                        const float* __restrict__ lnw,
                                        const float* __restrict__ lnb, int lane) {
  const int c = lane & 15;
  float bv[8], wv[8], sv[8];
#pragma unroll
  for (int t = 0; t < 8; ++t) {
    bv[t] = bias[t * 16 + c];
    wv[t] = lnw[t * 16 + c];
    sv[t] = lnb[t * 16 + c];
  }
#pragma unroll
  for (int t = 0; t < 8; ++t)
#pragma unroll
    for (int r = 0; r < 8; ++r) acc[t][r] += bv[t];

#pragma unroll
  for (int r = 0; r < 8; ++r) {
    float s = 0.0f, s2 = 0.0f;
#pragma unroll
    for (int t = 0; t < 8; ++t) {
      const float v = acc[t][r];
      s += v;
      s2 += v * v;
    }
#pragma unroll
    for (int m = 1; m < 16; m <<= 1) {
      s  += __shfl_xor(s, m, 32);
      s2 += __shfl_xor(s2, m, 32);
    }
    const float mu  = s * (1.0f / 128.0f);
    const float var = s2 * (1.0f / 128.0f) - mu * mu;
    const float rs  = rsqrtf(var + 1e-5f);
#pragma unroll
    for (int t = 0; t < 8; ++t) {
      const float v = (acc[t][r] - mu) * rs * wv[t] + sv[t];
      acc[t][r] = v * (1.0f / (1.0f + __expf(-v)));  // SiLU
    }
  }
}

// ---- write activations back to the LDS tile as bf16 (row-major) for the next GEMM.
__device__ __forceinline__ void store_act_bf16(__bf16* dst, const v8f acc[8], int rowBase, int lane) {
  const int c = lane & 15;
  const int half = lane >> 4;
#pragma unroll
  for (int r = 0; r < 8; ++r) {
    __bf16* row = dst + (rowBase + r + 8 * half) * HDIM + c;
#pragma unroll
    for (int t = 0; t < 8; ++t) row[t * 16] = (__bf16)acc[t][r];
  }
}

__global__ void __launch_bounds__(256, 1) fused_radial_mlp(
    const float* __restrict__ x,    const float* __restrict__ w1,
    const float* __restrict__ b1,   const float* __restrict__ ln1w,
    const float* __restrict__ ln1b, const float* __restrict__ w2,
    const float* __restrict__ b2,   const float* __restrict__ ln2w,
    const float* __restrict__ ln2b, const float* __restrict__ w3,
    const float* __restrict__ b3,   float* __restrict__ out) {
  // 64 KB LDS: two bf16 tiles during compute; overlaid fp32 output staging at the end.
  __shared__ __align__(16) __bf16 smem[2 * HDIM * HDIM];
  __bf16* sAct = smem;                 // 32 KB activation tile
  __bf16* sW   = smem + HDIM * HDIM;   // 32 KB weight tile
  float*  sOut = (float*)smem;         // 64 KB fp32 staging (after final GEMM)

  const int tid     = threadIdx.x;
  const int lane    = tid & 31;
  const int wave    = tid >> 5;
  const int n       = blockIdx.y;          // branch
  const int eBase   = blockIdx.x * HDIM;   // e-tile
  const int rowBase = wave * 16;
  const size_t wOff = (size_t)n * HDIM * HDIM;

  // Hint the later-stage weights into cache while stage 1 runs (global_prefetch_b8).
  {
    const char* p2 = (const char*)(w2 + wOff) + tid * 256;
    const char* p3 = (const char*)(w3 + wOff) + tid * 256;
    __builtin_prefetch(p2, 0, 1);
    __builtin_prefetch(p3, 0, 1);
  }

  load_tile_bf16(sAct, x + (size_t)eBase * HDIM, tid);
  load_tile_bf16(sW, w1 + wOff, tid);
  __syncthreads();

  v8f acc[8];

  // ---- stage 1: h = x @ W1[n]^T + b1 ; LN ; SiLU
  gemm128(acc, sAct, sW, rowBase, lane);
  __syncthreads();  // all waves done reading sAct/sW
  ln_silu(acc, b1 + n * HDIM, ln1w + n * HDIM, ln1b + n * HDIM, lane);
  store_act_bf16(sAct, acc, rowBase, lane);
  __syncthreads();
  load_tile_bf16(sW, w2 + wOff, tid);
  __syncthreads();

  // ---- stage 2: h = a @ fc2[n]^T + b2 ; LN ; SiLU
  gemm128(acc, sAct, sW, rowBase, lane);
  __syncthreads();
  ln_silu(acc, b2 + n * HDIM, ln2w + n * HDIM, ln2b + n * HDIM, lane);
  store_act_bf16(sAct, acc, rowBase, lane);
  __syncthreads();
  load_tile_bf16(sW, w3 + wOff, tid);
  __syncthreads();

  // ---- stage 3: out = a @ fc3[n]^T + b3
  gemm128(acc, sAct, sW, rowBase, lane);
  __syncthreads();  // sAct/sW dead from here: safe to overlay sOut

  const int c = lane & 15;
  const int half = lane >> 4;
  float bf[8];
#pragma unroll
  for (int t = 0; t < 8; ++t) bf[t] = b3[n * HDIM + t * 16 + c];

  // Scatter accumulators (+bias) into the LDS fp32 staging tile.
#pragma unroll
  for (int r = 0; r < 8; ++r) {
    float* row = sOut + (rowBase + r + 8 * half) * HDIM + c;
#pragma unroll
    for (int t = 0; t < 8; ++t) row[t * 16] = acc[t][r] + bf[t];
  }
  __syncthreads();

  // Stream the tile to HBM with fully coalesced 128-bit stores
  // (256 threads x float4 = 4 KB contiguous per iteration).
  float* __restrict__ outTile = out + ((size_t)n * E_TOT + (size_t)eBase) * HDIM;
#pragma unroll
  for (int it = 0; it < 16; ++it) {
    const int i = (it * 256 + tid) * 4;
    *(float4*)(outTile + i) = *(const float4*)(sOut + i);
  }
}

extern "C" void kernel_launch(void* const* d_in, const int* in_sizes, int n_in,
                              void* d_out, int out_size, void* d_ws, size_t ws_size,
                              hipStream_t stream) {
  (void)in_sizes; (void)n_in; (void)out_size; (void)d_ws; (void)ws_size;
  const float* x    = (const float*)d_in[0];
  const float* w1   = (const float*)d_in[1];
  const float* b1   = (const float*)d_in[2];
  const float* ln1w = (const float*)d_in[3];
  const float* ln1b = (const float*)d_in[4];
  const float* w2   = (const float*)d_in[5];
  const float* b2   = (const float*)d_in[6];
  const float* ln2w = (const float*)d_in[7];
  const float* ln2b = (const float*)d_in[8];
  const float* w3   = (const float*)d_in[9];
  const float* b3   = (const float*)d_in[10];
  float* out = (float*)d_out;

  dim3 grid(E_TOT / HDIM, 8, 1);  // 512 e-tiles x 8 branches
  dim3 block(256, 1, 1);          // 8 waves (wave32)
  fused_radial_mlp<<<grid, block, 0, stream>>>(x, w1, b1, ln1w, ln1b,
                                               w2, b2, ln2w, ln2b, w3, b3, out);
}
